// Corr1d_V2_21801253995202
// MI455X (gfx1250) — compile-verified, matrix operand order checked by reading, not run on hardware
//
#include <hip/hip_runtime.h>

// CDNA5 / gfx1250: wave32, WMMA 16x16x32 bf16 (f32 accum).
typedef __attribute__((ext_vector_type(16))) __bf16 v16bf;
typedef __attribute__((ext_vector_type(8)))  float  v8f;

#define Hh   256
#define Ww   512
#define Cc   64
#define DMAX 48
#define CSTRIDE (Hh * Ww)          // 131072 elements between channels
#define BSTRIDE (Cc * Hh * Ww)     // elements between batches (inputs)
#define LDS_PITCH 18               // even (float2 readout) + conflict-free scatter
#define TRASH_IDX (DMAX * LDS_PITCH)   // sink slot for invalid diagonal elements
#define WAVE_LDS  (DMAX * LDS_PITCH + 2)
#define WAVES_PER_BLOCK 8

__global__ __launch_bounds__(256)
void corr1d_wmma_kernel(const float* __restrict__ fL,
                        const float* __restrict__ fR,
                        float* __restrict__ out) {
    __shared__ float lds[WAVES_PER_BLOCK][WAVE_LDS];

    const int lane = threadIdx.x & 31;
    const int wv   = threadIdx.x >> 5;
    const int tile = blockIdx.x * WAVES_PER_BLOCK + wv;   // 0..32767

    const int xt = tile & 31;             // 32 x-tiles of 16
    const int h  = (tile >> 5) & (Hh - 1);
    const int b  = tile >> 13;            // 4 batches
    const int x0 = xt << 4;

    const size_t rowBase = (size_t)b * BSTRIDE + (size_t)h * Ww;

    // WMMA 16-bit A/B operand layout (16x32, MxK):
    //   lanes 0-15 : m(=n) = lane,   K = {0..7, 16..23}
    //   lanes 16-31: m(=n) = lane-16, K = {8..15, 24..31}
    const int nm    = lane & 15;               // m for A-tile, n for B-tile
    const int kbase = (lane < 16) ? 0 : 8;

    // ---- Load A: fL columns x0..x0+15, K=64 as two K=32 chunks, cvt->bf16 ----
    v16bf A[2];
    {
        const float* pL = fL + rowBase + (x0 + nm);
        #pragma unroll
        for (int kc = 0; kc < 2; ++kc) {
            #pragma unroll
            for (int j = 0; j < 16; ++j) {
                const int k = kc * 32 + kbase + j + ((j & 8) ? 8 : 0);
                A[kc][j] = (__bf16)pL[(size_t)k * CSTRIDE];
            }
        }
    }

    const int mhi = (lane < 16) ? 0 : 8;       // C/D layout: m = r + mhi, n = lane&15

    // ---- 4 output tiles along the disparity band ----
    #pragma unroll
    for (int tt = 0; tt < 4; ++tt) {
        const int xr  = x0 - 48 + tt * 16 + nm;     // R column for this lane
        const int xrc = (xr < 0) ? 0 : xr;          // clamp OOB reads (zeroed later)

        // Load B tile: fR columns, same per-lane K gather as A
        v16bf B[2];
        {
            const float* pR = fR + rowBase + xrc;
            #pragma unroll
            for (int kc = 0; kc < 2; ++kc) {
                #pragma unroll
                for (int j = 0; j < 16; ++j) {
                    const int k = kc * 32 + kbase + j + ((j & 8) ? 8 : 0);
                    B[kc][j] = (__bf16)pR[(size_t)k * CSTRIDE];
                }
            }
        }

        // D = A * B + C : chain two K=32 bf16 WMMAs for K=64
        v8f c = {};
        c = __builtin_amdgcn_wmma_f32_16x16x32_bf16(false, A[0], false, B[0],
                                                    (short)0, c, false, false);
        c = __builtin_amdgcn_wmma_f32_16x16x32_bf16(false, A[1], false, B[1],
                                                    (short)0, c, false, false);

        // Branch-free scatter of the valid diagonals into (d, x) LDS tile:
        // invalid elements go to a per-wave trash slot via cndmask'd address.
        #pragma unroll
        for (int r = 0; r < 8; ++r) {
            const int m = r + mhi;                     // x index within tile
            const int d = (x0 + m) - xr;               // disparity of this element
            const bool valid = ((unsigned)d < (unsigned)DMAX);
            const float v = (xr >= 0) ? c[r] : 0.0f;   // mask x < d
            const int idx = valid ? (d * LDS_PITCH + m) : TRASH_IDX;
            lds[wv][idx] = v;
        }
    }

    __syncthreads();   // per-wave LDS slices; block barrier keeps it simple

    // ---- Coalesced float2 writeout: 4 disparity rows (4 x 64B) per iteration ----
    #pragma unroll
    for (int it = 0; it < DMAX / 4; ++it) {
        const int d  = it * 4 + (lane >> 3);
        const int m0 = (lane & 7) * 2;
        const float2 v = *(const float2*)&lds[wv][d * LDS_PITCH + m0];
        *(float2*)&out[((size_t)(b * DMAX + d) * Hh + h) * Ww + x0 + m0] = v;
    }
}

extern "C" void kernel_launch(void* const* d_in, const int* in_sizes, int n_in,
                              void* d_out, int out_size, void* d_ws, size_t ws_size,
                              hipStream_t stream) {
    (void)in_sizes; (void)n_in; (void)d_ws; (void)ws_size; (void)out_size;
    const float* fL = (const float*)d_in[0];
    const float* fR = (const float*)d_in[1];
    float* out = (float*)d_out;

    // tiles = b(4) * h(256) * x-tiles(32) = 32768 waves; 8 waves per block
    const int total_waves = 4 * Hh * (Ww / 16);
    const int blocks = total_waves / WAVES_PER_BLOCK;   // 4096
    corr1d_wmma_kernel<<<blocks, WAVES_PER_BLOCK * 32, 0, stream>>>(fL, fR, out);
}
